// GISSA_26508538151242
// MI455X (gfx1250) — matrix operand order repaired and checked by reading, not compiled, not run on hardware
//
#include <hip/hip_runtime.h>

// GISSA fused pipeline for MI455X (gfx1250, wave32, WMMA + TDM).
// K1: per (b,h) Gram Xq·Xk^T via v_wmma_f32_16x16x32_bf16, then A = signed-sqrt-softmax(scale*(Wq G Wk^T + rank-1 bias terms)),
//     M = A*Wv (bf16), c = A*bv.
// K2: out = M @ Xv (WMMA) + c + x residual, channel permute, BN -> y (f32, staged in d_out) and relu(y) as bf16 -> ws.
// K3: per (b,d) 8x8 stage-2 Grams + attention -> M2, c2.
// K4: x2 = M2 @ xr + c2 + y, updating d_out in place; xr tile staged to LDS via Tensor Data Mover
//     (tensor_load_to_lds + s_wait_tensorcnt), overlapped with parameter loads.

#define Bn  16
#define Cn  512
#define NHn 8
#define HDn 64
#define Nn  4096

typedef __attribute__((ext_vector_type(16))) __bf16 v16bf;
typedef __attribute__((ext_vector_type(8)))  float  v8f;
typedef __attribute__((ext_vector_type(4)))  unsigned u32x4;
typedef __attribute__((ext_vector_type(8)))  int      i32x8;
typedef __attribute__((ext_vector_type(4)))  int      i32x4;

// ---- WMMA fragment loaders (contiguous-K order, same for A and B) ----
// A (16x32 bf16): lane = M row, 16 contiguous K per lane (K-base = 16 * lane-half)
__device__ __forceinline__ v16bf afrag(const __bf16* base, int stride, int m0, int kc, int lane) {
  const int m  = m0 + (lane & 15);
  const int hh = (lane >> 4) << 4;
  const __bf16* p = base + m * stride + kc;
  v16bf a;
#pragma unroll
  for (int e = 0; e < 16; ++e) a[e] = p[hh + e];
  return a;
}
// B (32x16 bf16): lane = N column, 16 contiguous K per lane (K-base = 16 * lane-half)
__device__ __forceinline__ v16bf bfrag(const __bf16* base, int stride, int n0, int kc, int lane) {
  const int nc = n0 + (lane & 15);
  const int kb = kc + ((lane >> 4) << 4);
  const __bf16* p = base + nc * stride + kb;
  v16bf bb;
#pragma unroll
  for (int e = 0; e < 16; ++e) bb[e] = p[e];
  return bb;
}

__device__ __forceinline__ float ssqrt(float a) {
  float s = (a > 0.f) ? 1.f : ((a < 0.f) ? -1.f : 0.f);
  return s * sqrtf(fabsf(a) + 1e-5f);
}

// =================== K1: stage-1 attention (Gram trick) ===================
__global__ __launch_bounds__(512) void k1_stage1_attn(
    const float* __restrict__ x, const float* __restrict__ w1, const float* __restrict__ b1,
    __bf16* __restrict__ M1, float* __restrict__ c1)
{
  __shared__ __bf16 sXq[64][144];
  __shared__ __bf16 sXk[64][144];
  __shared__ float Gs[64][65];
  __shared__ float Ws[64][65];
  __shared__ float T1[64][65];
  __shared__ float As[64][65];
  __shared__ float sq_[64], sk_[64], uq[64], vk[64], bqv[64], bkv[64], bvv[64];

  const int tid = threadIdx.x, lane = tid & 31, wv = tid >> 5;
  const int mi = wv >> 2, ni = wv & 3;
  const int b = blockIdx.x >> 3, h = blockIdx.x & 7;
  const int cq = h*HDn, ck = Cn + h*HDn, cv = 2*Cn + h*HDn;
  const int gq = cq/192, oq = cq - gq*192;   // each 64-ch slice lies inside one conv group
  const int gk = ck/192, ok = ck - gk*192;
  const int gv = cv/192, ov = cv - gv*192;
  const float* xq = x + ((size_t)b*Cn + gq*HDn) * Nn;
  const float* xk = x + ((size_t)b*Cn + gk*HDn) * Nn;

  if (tid < 64) { sq_[tid] = 0.f; sk_[tid] = 0.f; }

  v8f acc = {0.f,0.f,0.f,0.f,0.f,0.f,0.f,0.f};

  // ---- G = Xq * Xk^T over N=4096, 32 stages of 128 columns ----
  for (int s = 0; s < 32; ++s) {
    const int n0 = s * 128;
    __syncthreads();
    for (int t = tid; t < 2048; t += 512) {
      const int row = t >> 5, c4 = (t & 31) << 2;
      float4 fq = *(const float4*)(xq + (size_t)row*Nn + n0 + c4);
      float4 fk = *(const float4*)(xk + (size_t)row*Nn + n0 + c4);
      sXq[row][c4+0] = (__bf16)fq.x; sXq[row][c4+1] = (__bf16)fq.y;
      sXq[row][c4+2] = (__bf16)fq.z; sXq[row][c4+3] = (__bf16)fq.w;
      sXk[row][c4+0] = (__bf16)fk.x; sXk[row][c4+1] = (__bf16)fk.y;
      sXk[row][c4+2] = (__bf16)fk.z; sXk[row][c4+3] = (__bf16)fk.w;
      atomicAdd(&sq_[row], fq.x+fq.y+fq.z+fq.w);
      atomicAdd(&sk_[row], fk.x+fk.y+fk.z+fk.w);
    }
    __syncthreads();
#pragma unroll
    for (int kc = 0; kc < 128; kc += 32) {
      v16bf a  = afrag(&sXq[0][0], 144, mi*16, kc, lane);
      v16bf bb = bfrag(&sXk[0][0], 144, ni*16, kc, lane);
      acc = __builtin_amdgcn_wmma_f32_16x16x32_bf16(false, a, false, bb, (short)0, acc, false, false);
    }
  }
  { // spill accumulators to Gs per C-layout (VGPR r -> M = mtile + 8*half + r, N = lane&15)
    const int nn = ni*16 + (lane & 15);
    const int mb = mi*16 + ((lane >> 4) << 3);
#pragma unroll
    for (int r = 0; r < 8; ++r) Gs[mb + r][nn] = acc[r];
  }
  // ---- Phase A: T1 = Wq * G ; uq = Wq * rowsum(Xq) ----
  for (int t = tid; t < 4096; t += 512) {
    const int d = t >> 6, i = t & 63;
    Ws[d][i] = w1[((size_t)gq*192 + oq + d)*64 + i];
  }
  if (tid < 64) bqv[tid] = b1[gq*192 + oq + tid];
  __syncthreads();
  for (int t = tid; t < 4096; t += 512) {
    const int d = t >> 6, e = t & 63;
    float sacc = 0.f;
    for (int j = 0; j < 64; ++j) sacc += Ws[d][j] * Gs[j][e];
    T1[d][e] = sacc;
  }
  if (tid < 64) {
    float sacc = 0.f;
    for (int j = 0; j < 64; ++j) sacc += Ws[tid][j] * sq_[j];
    uq[tid] = sacc;
  }
  __syncthreads();
  // ---- Phase B: A = scale*(T1*Wk^T + uq*bk^T + bq*vk^T + N*bq*bk^T) ----
  for (int t = tid; t < 4096; t += 512) {
    const int d = t >> 6, i = t & 63;
    Ws[d][i] = w1[((size_t)gk*192 + ok + d)*64 + i];
  }
  if (tid < 64) bkv[tid] = b1[gk*192 + ok + tid];
  __syncthreads();
  if (tid < 64) {
    float sacc = 0.f;
    for (int j = 0; j < 64; ++j) sacc += Ws[tid][j] * sk_[j];
    vk[tid] = sacc;
  }
  __syncthreads();
  for (int t = tid; t < 4096; t += 512) {
    const int d = t >> 6, e = t & 63;
    float sacc = 0.f;
    for (int j = 0; j < 64; ++j) sacc += T1[d][j] * Ws[e][j];
    As[d][e] = 0.125f * (sacc + uq[d]*bkv[e] + bqv[d]*vk[e] + 4096.f*bqv[d]*bkv[e]);
  }
  __syncthreads();
  // ---- signed-sqrt softmax, one row per thread ----
  if (tid < 64) {
    float mx = -3.4e38f;
    for (int e = 0; e < 64; ++e) { float a = ssqrt(As[tid][e]); As[tid][e] = a; mx = fmaxf(mx, a); }
    float sum = 0.f;
    for (int e = 0; e < 64; ++e) { float ev = __expf(As[tid][e] - mx); As[tid][e] = ev; sum += ev; }
    const float inv = 1.f / sum;
    for (int e = 0; e < 64; ++e) As[tid][e] *= inv;
  }
  __syncthreads();
  // ---- Phase C: M = A*Wv (store bf16), c = A*bv ----
  for (int t = tid; t < 4096; t += 512) {
    const int d = t >> 6, i = t & 63;
    Ws[d][i] = w1[((size_t)gv*192 + ov + d)*64 + i];
  }
  if (tid < 64) bvv[tid] = b1[gv*192 + ov + tid];
  __syncthreads();
  for (int t = tid; t < 4096; t += 512) {
    const int d = t >> 6, i = t & 63;
    float sacc = 0.f;
    for (int e = 0; e < 64; ++e) sacc += As[d][e] * Ws[e][i];
    M1[(((size_t)b*NHn + h)*64 + d)*64 + i] = (__bf16)sacc;
  }
  if (tid < 64) {
    float sacc = 0.f;
    for (int e = 0; e < 64; ++e) sacc += As[tid][e] * bvv[e];
    c1[((size_t)b*NHn + h)*64 + tid] = sacc;
  }
}

// =================== K2: out = M@Xv + c + x, permute, BN, ReLU ===================
__global__ __launch_bounds__(512) void k2_stage1_out(
    const float* __restrict__ x, const __bf16* __restrict__ M1, const float* __restrict__ c1,
    const float* __restrict__ gamma, const float* __restrict__ beta,
    const float* __restrict__ rmean, const float* __restrict__ rvar,
    float* __restrict__ y_out, __bf16* __restrict__ xr)
{
  __shared__ __bf16 sM[64][72];     // M row-major [d][i]
  __shared__ __bf16 sXT[256][80];   // Xv transposed [n][i] -> contiguous-K B-frags
  __shared__ float sInv[64], sBias[64], sC[64];

  const int tid = threadIdx.x, lane = tid & 31, wv = tid >> 5;
  const int nt = blockIdx.x & 15;
  const int h  = (blockIdx.x >> 4) & 7;
  const int b  = blockIdx.x >> 7;
  const int n0 = nt * 256;
  const int gv = (2*Cn + h*HDn) / 192;

  for (int t = tid; t < 4096; t += 512) {
    const int d = t >> 6, i = t & 63;
    sM[d][i] = M1[(((size_t)b*NHn + h)*64 + d)*64 + i];
  }
  const float* xv = x + ((size_t)b*Cn + gv*HDn) * Nn;
  for (int t = tid; t < 4096; t += 512) {
    const int i = t >> 6, c4 = (t & 63) << 2;
    float4 f = *(const float4*)(xv + (size_t)i*Nn + n0 + c4);
    sXT[c4+0][i] = (__bf16)f.x; sXT[c4+1][i] = (__bf16)f.y;
    sXT[c4+2][i] = (__bf16)f.z; sXT[c4+3][i] = (__bf16)f.w;
  }
  if (tid < 64) {
    const int cp = tid*8 + h;                 // permuted channel
    const float iv = gamma[cp] * rsqrtf(rvar[cp] + 1e-5f);
    sInv[tid]  = iv;
    sBias[tid] = beta[cp] - rmean[cp]*iv;
    sC[tid]    = c1[((size_t)b*NHn + h)*64 + tid];
    __builtin_prefetch(x + ((size_t)b*Cn + h*64 + tid)*Nn + n0, 0, 1); // residual stream
  }
  __syncthreads();

  const int m  = wv >> 2;
  const int jb = (wv & 3) << 2;
  v8f z = {0.f,0.f,0.f,0.f,0.f,0.f,0.f,0.f};
  v8f acc[4] = {z, z, z, z};
#pragma unroll
  for (int kc = 0; kc < 64; kc += 32) {
    v16bf a = afrag(&sM[0][0], 72, m*16, kc, lane);
#pragma unroll
    for (int p = 0; p < 4; ++p) {
      v16bf bb = bfrag(&sXT[0][0], 80, (jb + p)*16, kc, lane);
      acc[p] = __builtin_amdgcn_wmma_f32_16x16x32_bf16(false, a, false, bb, (short)0, acc[p], false, false);
    }
  }
  const float* xres = x + ((size_t)b*Cn + h*64) * Nn;
#pragma unroll
  for (int p = 0; p < 4; ++p) {
    const int nn = n0 + (jb + p)*16 + (lane & 15);
    const int mb = m*16 + ((lane >> 4) << 3);
#pragma unroll
    for (int r = 0; r < 8; ++r) {
      const int d = mb + r;
      const float v  = acc[p][r] + sC[d] + xres[(size_t)d*Nn + nn];
      const float yv = v * sInv[d] + sBias[d];
      const size_t o = ((size_t)b*Cn + (size_t)(d*8 + h))*Nn + nn;
      y_out[o] = yv;
      xr[o] = (__bf16)fmaxf(yv, 0.f);
    }
  }
}

// =================== K3: stage-2 8x8 attention ===================
__global__ __launch_bounds__(256) void k3_stage2_attn(
    const __bf16* __restrict__ xr, const float* __restrict__ w2, const float* __restrict__ b2,
    float* __restrict__ M2, float* __restrict__ c2)
{
  __shared__ float Gs2[8][9];
  __shared__ float sas[8], sbs[8];
  __shared__ float A2s[8][9];

  const int tid = threadIdx.x, lane = tid & 31;
  const int d = blockIdx.x & 63, b = blockIdx.x >> 6;
  const int cq = 8*d, ck = Cn + 8*d, cv = 2*Cn + 8*d;
  const int gq = cq/24, oq = cq - gq*24;
  const int gk = ck/24, ok = ck - gk*24;
  const int gv = cv/24, ov = cv - gv*24;
  const __bf16* xa = xr + ((size_t)b*Cn + gq*8) * Nn;
  const __bf16* xb = xr + ((size_t)b*Cn + gk*8) * Nn;

  float g[8][8], sa[8], sb[8];
#pragma unroll
  for (int i = 0; i < 8; ++i) { sa[i]=0.f; sb[i]=0.f;
#pragma unroll
    for (int j = 0; j < 8; ++j) g[i][j]=0.f; }

  for (int n = tid; n < Nn; n += 256) {
    float av[8], bw[8];
#pragma unroll
    for (int i = 0; i < 8; ++i) { av[i] = (float)xa[(size_t)i*Nn + n]; bw[i] = (float)xb[(size_t)i*Nn + n]; }
#pragma unroll
    for (int i = 0; i < 8; ++i) {
      sa[i] += av[i]; sb[i] += bw[i];
#pragma unroll
      for (int j = 0; j < 8; ++j) g[i][j] += av[i]*bw[j];
    }
  }
  if (tid < 64) Gs2[tid>>3][tid&7] = 0.f;
  if (tid < 8) { sas[tid]=0.f; sbs[tid]=0.f; }
  __syncthreads();
#pragma unroll
  for (int i = 0; i < 8; ++i) {
#pragma unroll
    for (int j = 0; j < 8; ++j) {
      float v = g[i][j];
      for (int off = 16; off > 0; off >>= 1) v += __shfl_down(v, off);
      if (lane == 0) atomicAdd(&Gs2[i][j], v);
    }
    float va = sa[i];
    for (int off = 16; off > 0; off >>= 1) va += __shfl_down(va, off);
    if (lane == 0) atomicAdd(&sas[i], va);
    float vb = sb[i];
    for (int off = 16; off > 0; off >>= 1) vb += __shfl_down(vb, off);
    if (lane == 0) atomicAdd(&sbs[i], vb);
  }
  __syncthreads();
  if (tid < 64) {
    const int hh = tid >> 3, e = tid & 7;
    float accv = 0.f, uq2 = 0.f, vk2 = 0.f;
#pragma unroll
    for (int i = 0; i < 8; ++i) {
      const float wq = w2[((size_t)gq*24 + oq + hh)*8 + i];
      float t = 0.f;
#pragma unroll
      for (int j = 0; j < 8; ++j) t += Gs2[i][j] * w2[((size_t)gk*24 + ok + e)*8 + j];
      accv += wq * t;
      uq2 += wq * sas[i];
      vk2 += w2[((size_t)gk*24 + ok + e)*8 + i] * sbs[i];
    }
    const float bq2 = b2[gq*24 + oq + hh], bk2 = b2[gk*24 + ok + e];
    A2s[hh][e] = ssqrt(0.35355339059f * (accv + uq2*bk2 + bq2*vk2 + 4096.f*bq2*bk2));
  }
  __syncthreads();
  if (tid < 8) {
    float mx = -3.4e38f;
    for (int e = 0; e < 8; ++e) mx = fmaxf(mx, A2s[tid][e]);
    float sum = 0.f;
    for (int e = 0; e < 8; ++e) { float ev = __expf(A2s[tid][e]-mx); A2s[tid][e]=ev; sum+=ev; }
    const float inv = 1.f/sum;
    for (int e = 0; e < 8; ++e) A2s[tid][e] *= inv;
  }
  __syncthreads();
  if (tid < 64) {
    const int hh = tid >> 3, i = tid & 7;
    float sacc = 0.f;
#pragma unroll
    for (int e = 0; e < 8; ++e) sacc += A2s[hh][e] * w2[((size_t)gv*24 + ov + e)*8 + i];
    M2[(((size_t)b*64 + d)*8 + hh)*8 + i] = sacc;
  }
  if (tid < 8) {
    float sacc = 0.f;
#pragma unroll
    for (int e = 0; e < 8; ++e) sacc += A2s[tid][e] * b2[gv*24 + ov + e];
    c2[((size_t)b*64 + d)*8 + tid] = sacc;
  }
}

// =================== K4: x2 = M2 @ xr + c2 + y (in-place on d_out) ===================
// xr tile (176 rows x 256 cols bf16, row stride 4096 elems) staged to LDS by the
// Tensor Data Mover: one 2-D D# descriptor, rows land contiguously (256*2B = 512B)
// exactly matching sXr's layout. Wave 0 issues + waits TENSORcnt; DMA overlaps the
// M2/c2 parameter loads, then the workgroup barrier publishes the tile.
__global__ __launch_bounds__(256) void k4_final(
    const __bf16* __restrict__ xr, const float* __restrict__ M2, const float* __restrict__ c2,
    float* __restrict__ io)
{
  __shared__ __bf16 sXr[176][256];   // xr channels 336..511 (all gv2 octets)
  __shared__ float sM2[64][8];
  __shared__ float sC2[64];

  const int tid = threadIdx.x, wv = tid >> 5;
  const int nt = blockIdx.x & 15;
  const int h  = (blockIdx.x >> 4) & 7;
  const int b  = blockIdx.x >> 7;
  const int n0 = nt * 256;

#if defined(__has_builtin) && __has_builtin(__builtin_amdgcn_tensor_load_to_lds)
  if (wv == 0) {
    const unsigned lds_off = (unsigned)(uintptr_t)&sXr[0][0];     // flat addr low 32 = LDS offset
    const unsigned long long ga =
        (unsigned long long)(uintptr_t)(xr + ((size_t)b*Cn + 336)*Nn + n0);
    // D# group 0: count=1 | lds_addr | global_addr[56:0] | type=2 ("image")
    u32x4 g0 = { 1u,
                 lds_off,
                 (unsigned)(ga & 0xffffffffu),
                 (unsigned)((ga >> 32) & 0x01ffffffu) | 0x80000000u };
    // D# group 1: data_size=1 (2B) | tensor_dim0=4096 | tensor_dim1=176 |
    //             tile_dim0=256 | tile_dim1=176 | tensor_dim0_stride=4096
    i32x8 g1 = { (int)0x00010000,
                 (int)(4096u << 16),
                 (int)(176u  << 16),
                 (int)(256u  << 16),
                 (int)176,
                 (int)4096,
                 0, 0 };
    i32x4 gz4 = {0, 0, 0, 0};
    i32x8 gz8 = {0, 0, 0, 0, 0, 0, 0, 0};
    // 6-arg variant (this toolchain): (g0, g1, g2, g3, g4, cpol)
    __builtin_amdgcn_tensor_load_to_lds(g0, g1, gz4, gz4, gz8, 0);
  }
#else
  for (int t = tid; t < 176*64; t += 256) {
    const int row = t >> 6, c4 = (t & 63) << 2;
    const size_t go = ((size_t)b*Cn + 336 + row)*Nn + n0 + c4;
    *(uint2*)&sXr[row][c4] = *(const uint2*)&xr[go];   // 4x bf16
  }
#endif

  for (int t = tid; t < 512; t += 256) {
    const int dd = t >> 3, i = t & 7;
    sM2[dd][i] = M2[(((size_t)b*64 + dd)*8 + h)*8 + i];
  }
  if (tid < 64) sC2[tid] = c2[((size_t)b*64 + tid)*8 + h];

#if defined(__has_builtin) && __has_builtin(__builtin_amdgcn_tensor_load_to_lds)
  if (wv == 0) __builtin_amdgcn_s_wait_tensorcnt(0);   // tile in LDS before barrier signal
#endif
  __syncthreads();

  const size_t basey = ((size_t)b*Cn + (size_t)h*64)*Nn + n0 + tid;
#pragma unroll 4
  for (int dd = 0; dd < 64; ++dd) {
    const int rb = ((2*Cn + 8*dd)/24)*8 - 336;
    float s = sC2[dd];
#pragma unroll
    for (int i = 0; i < 8; ++i) s += sM2[dd][i] * (float)sXr[rb + i][tid];
    const size_t o = basey + (size_t)dd*Nn;
    io[o] += s;   // read y, write x2 (same address, block-local)
  }
}

// =================== launch ===================
extern "C" void kernel_launch(void* const* d_in, const int* in_sizes, int n_in,
                              void* d_out, int out_size, void* d_ws, size_t ws_size,
                              hipStream_t stream) {
  (void)in_sizes; (void)n_in; (void)out_size; (void)ws_size;
  const float* x     = (const float*)d_in[0];
  const float* w1    = (const float*)d_in[1];
  const float* b1    = (const float*)d_in[2];
  const float* w2    = (const float*)d_in[3];
  const float* b2    = (const float*)d_in[4];
  const float* gamma = (const float*)d_in[5];
  const float* beta  = (const float*)d_in[6];
  const float* rmean = (const float*)d_in[7];
  const float* rvar  = (const float*)d_in[8];
  float* out = (float*)d_out;

  char* ws = (char*)d_ws;
  constexpr size_t OFF_M1 = 0;                       // 16*8*64*64 bf16   = 1 MiB
  constexpr size_t OFF_C1 = OFF_M1 + 1048576;        // 16*8*64 f32
  constexpr size_t OFF_M2 = OFF_C1 + 32768;          // 16*64*64 f32
  constexpr size_t OFF_C2 = OFF_M2 + 262144;         // 16*64*8 f32
  constexpr size_t OFF_XR = OFF_C2 + 32768;          // 16*512*4096 bf16  = 64 MiB
  __bf16* M1 = (__bf16*)(ws + OFF_M1);
  float*  c1 = (float*)(ws + OFF_C1);
  float*  M2 = (float*)(ws + OFF_M2);
  float*  c2 = (float*)(ws + OFF_C2);
  __bf16* xr = (__bf16*)(ws + OFF_XR);

  k1_stage1_attn<<<dim3(Bn*NHn), dim3(512), 0, stream>>>(x, w1, b1, M1, c1);
  k2_stage1_out <<<dim3(Bn*NHn*16), dim3(512), 0, stream>>>(x, M1, c1, gamma, beta, rmean, rvar, out, xr);
  k3_stage2_attn<<<dim3(Bn*64), dim3(256), 0, stream>>>(xr, w2, b2, M2, c2);
  k4_final      <<<dim3(Bn*NHn*16), dim3(256), 0, stream>>>(xr, M2, c2, out);
}
